// ExtensibleAttention_single_69836168233356
// MI455X (gfx1250) — matrix operand order repaired and checked by reading, not compiled, not run on hardware
//
#include <hip/hip_runtime.h>
#include <math.h>
#include <stdint.h>

#define C_DIM 256
#define HEADS 8
#define HDIM  32

typedef __attribute__((ext_vector_type(8)))  __bf16 v8bf;
typedef __attribute__((ext_vector_type(16))) __bf16 v16bf;
typedef __attribute__((ext_vector_type(8)))  float  v8f;

// --------------------------------------------------------------------------
// Async global->LDS path (CDNA5 GLOBAL_LOAD_ASYNC_TO_LDS_B128, ASYNCcnt).
// clang-22 signature: (v4i addrspace(1)*, v4i addrspace(3)*, imm, imm).
// Guarded: falls back to load+ds_store if the builtins are not available.
// --------------------------------------------------------------------------
#if defined(__has_builtin)
#if __has_builtin(__builtin_amdgcn_global_load_async_to_lds_b128) && \
    __has_builtin(__builtin_amdgcn_s_wait_asynccnt)
#define USE_ASYNC_LDS 1
#endif
#endif
#ifndef USE_ASYNC_LDS
#define USE_ASYNC_LDS 0
#endif

#if USE_ASYNC_LDS
typedef int v4i_t __attribute__((vector_size(16)));
typedef v4i_t __attribute__((address_space(1))) as1_v4i;
typedef v4i_t __attribute__((address_space(3))) as3_v4i;
// flat LDS pointers carry the LDS byte offset in their low 32 bits, so an
// integer round-trip gives us valid AS(1)/AS(3) pointers for the builtin.
#define GPTR(p) ((as1_v4i*)(unsigned long long)(uintptr_t)(p))
#define LPTR(p) ((as3_v4i*)(unsigned int)(uintptr_t)(p))
#endif

static __device__ __forceinline__ unsigned short f32_to_bf16_bits(float f) {
  union { float f; unsigned int u; } cv; cv.f = f;
  unsigned int u = cv.u;
  u += 0x7FFFu + ((u >> 16) & 1u);   // round-to-nearest-even
  return (unsigned short)(u >> 16);
}

static __device__ __forceinline__ unsigned int pack_bf16x2(float lo, float hi) {
  return (unsigned int)f32_to_bf16_bits(lo) |
         ((unsigned int)f32_to_bf16_bits(hi) << 16);
}

static __device__ __forceinline__ v16bf cat8(v8bf lo, v8bf hi) {
  return __builtin_shufflevector(lo, hi, 0,1,2,3,4,5,6,7,8,9,10,11,12,13,14,15);
}

// --------------------------------------------------------------------------
// Vectorized f32 -> bf16 conversion: 8 elements / thread.
// --------------------------------------------------------------------------
__global__ __launch_bounds__(256)
void convert_f32_bf16(const float* __restrict__ src,
                      unsigned short* __restrict__ dst, int n8) {
  const int i = blockIdx.x * 256 + threadIdx.x;
  if (i < n8) {
    const float4 a = ((const float4*)src)[(size_t)i * 2 + 0];
    const float4 b = ((const float4*)src)[(size_t)i * 2 + 1];
    uint4 o;
    o.x = pack_bf16x2(a.x, a.y);
    o.y = pack_bf16x2(a.z, a.w);
    o.z = pack_bf16x2(b.x, b.y);
    o.w = pack_bf16x2(b.z, b.w);
    ((uint4*)dst)[i] = o;
  }
}

// --------------------------------------------------------------------------
// Out[M x 256] = A1 @ W1^T (+ A2 @ W2^T) + b1 (+ b2).
// All matrix inputs pre-converted to bf16. Block = 256 threads = 8 waves.
// Block tile 64(M) x 128(N); wave tile 32x32 -> 4 WMMA / K-step / wave.
// Double-buffered LDS staging; async global->LDS when available.
// --------------------------------------------------------------------------
__global__ __launch_bounds__(256)
void gemm_bias_wmma(const unsigned short* __restrict__ A1,
                    const unsigned short* __restrict__ W1,
                    const float* __restrict__ b1,
                    const unsigned short* __restrict__ A2,
                    const unsigned short* __restrict__ W2,
                    const float* __restrict__ b2,
                    float* __restrict__ Out, int M) {
  __shared__ unsigned short As[2][64][40];    // 64 rows x 32 k (+pad to 80B)
  __shared__ unsigned short Bs[2][128][40];   // 128 cols(n) x 32 k

  const int tid  = threadIdx.x;
  const int lane = tid & 31;
  const int w    = tid >> 5;
  const int wm   = w & 1;        // 2 M positions of 32 rows
  const int wn   = w >> 1;       // 4 N positions of 32 cols
  const int m0   = blockIdx.x * 64;
  const int n0   = blockIdx.y * 128;

  const int khalf = lane >> 4;   // 0/1 lane group
  const int lrow  = lane & 15;

  v8f acc[2][2] = {};

  // stage one 64x32 A tile + 128x32 B tile into LDS buffer `buf`
  auto stage = [&](const unsigned short* Ain, const unsigned short* Wmat,
                   int buf, int kb) {
    {
      const int r = tid >> 2, c = (tid & 3) << 3;   // 64 rows x 4 chunks
      const unsigned short* src = Ain + (size_t)(m0 + r) * C_DIM + kb + c;
#if USE_ASYNC_LDS
      __builtin_amdgcn_global_load_async_to_lds_b128(GPTR(src),
                                                     LPTR(&As[buf][r][c]), 0, 0);
#else
      const uint4 v = *(const uint4*)src;
      *(uint4*)&As[buf][r][c] = v;
#endif
    }
#pragma unroll
    for (int j = 0; j < 2; ++j) {                   // 128 rows x 4 chunks
      const int idx = tid + j * 256;
      const int r = idx >> 2, c = (idx & 3) << 3;
      const unsigned short* src = Wmat + (size_t)(n0 + r) * C_DIM + kb + c;
#if USE_ASYNC_LDS
      __builtin_amdgcn_global_load_async_to_lds_b128(GPTR(src),
                                                     LPTR(&Bs[buf][r][c]), 0, 0);
#else
      const uint4 v = *(const uint4*)src;
      *(uint4*)&Bs[buf][r][c] = v;
#endif
    }
  };

  const int npass = (A2 != nullptr) ? 2 : 1;
  for (int pass = 0; pass < npass; ++pass) {
    const unsigned short* Ain  = pass ? A2 : A1;
    const unsigned short* Wmat = pass ? W2 : W1;

    stage(Ain, Wmat, 0, 0);
#pragma unroll
    for (int kbi = 0; kbi < 8; ++kbi) {
      const int cur = kbi & 1;
      if (kbi < 7) stage(Ain, Wmat, cur ^ 1, (kbi + 1) * 32);
#if USE_ASYNC_LDS
      if (kbi < 7) __builtin_amdgcn_s_wait_asynccnt(3);  // next tile may fly
      else         __builtin_amdgcn_s_wait_asynccnt(0);  // drain
#endif
      __syncthreads();

      // A 16x32 frags: elems 0-7 -> K = 8*khalf + j ; 8-15 -> 16 + 8*khalf + j
      v16bf afrag[2], bfrag[2];
#pragma unroll
      for (int mf = 0; mf < 2; ++mf) {
        const int mrow = wm * 32 + mf * 16 + lrow;
        v8bf alo = *(const v8bf*)&As[cur][mrow][8 * khalf];
        v8bf ahi = *(const v8bf*)&As[cur][mrow][16 + 8 * khalf];
        afrag[mf] = cat8(alo, ahi);
      }
      // B 32x16 frags: elems 0-7 -> K = 8*khalf + j ; 8-15 -> 16 + 8*khalf + j
#pragma unroll
      for (int nf = 0; nf < 2; ++nf) {
        const int ncol = wn * 32 + nf * 16 + lrow;
        v8bf blo = *(const v8bf*)&Bs[cur][ncol][8 * khalf];
        v8bf bhi = *(const v8bf*)&Bs[cur][ncol][16 + 8 * khalf];
        bfrag[nf] = cat8(blo, bhi);
      }
#pragma unroll
      for (int mf = 0; mf < 2; ++mf)
#pragma unroll
        for (int nf = 0; nf < 2; ++nf)
          acc[mf][nf] = __builtin_amdgcn_wmma_f32_16x16x32_bf16(
              false, afrag[mf], false, bfrag[nf], (short)0, acc[mf][nf],
              false, false);
      __syncthreads();
    }
  }

  // epilogue: bias + store (C/D layout: VGPR r -> M = r + 8*(lane>>4))
#pragma unroll
  for (int nf = 0; nf < 2; ++nf) {
    const int gn = n0 + wn * 32 + nf * 16 + lrow;
    float bias = (b1 != nullptr) ? b1[gn] : 0.f;
    if (b2 != nullptr) bias += b2[gn];
#pragma unroll
    for (int mf = 0; mf < 2; ++mf) {
      const int mbase = m0 + wm * 32 + mf * 16 + 8 * khalf;
#pragma unroll
      for (int r = 0; r < 8; ++r)
        Out[(size_t)(mbase + r) * C_DIM + gn] = acc[mf][nf][r] + bias;
    }
  }
}

// --------------------------------------------------------------------------
// Per-location kernel: offset projection, bilinear gather of K/V, per-head
// dot, softmax over the 8 heads, weighted V (emitted directly as bf16).
// One block per (n,l); wave h = head h, lane d = channel (D==32==wave32).
// --------------------------------------------------------------------------
__global__ __launch_bounds__(256)
void attn_sample_kernel(const float* __restrict__ Q, const float* __restrict__ K,
                        const float* __restrict__ V,
                        const float* __restrict__ query,
                        const float* __restrict__ Woff,
                        const float* __restrict__ boff,
                        const float* __restrict__ refpts,
                        unsigned short* __restrict__ attn_bf,
                        const int* __restrict__ pHf,
                        const int* __restrict__ pWf) {
  const int Hf = pHf[0], Wf = pWf[0];
  const int L  = Hf * Wf;
  const int i  = blockIdx.x;          // flat (n, l)
  const int n  = i / L;
  const int tid = threadIdx.x;
  const int h  = tid >> 5;
  const int d  = tid & 31;

  __shared__ float qrow[C_DIM];
  __shared__ float alog[HEADS];

  qrow[tid] = query[(size_t)i * C_DIM + tid];
  __syncthreads();

  // per-head (dx, dy) offset projection: lane-strided dot + wave reduction
  float sx = 0.f, sy = 0.f;
  const float* wrx = Woff + (size_t)(2 * h) * C_DIM;
  const float* wry = Woff + (size_t)(2 * h + 1) * C_DIM;
#pragma unroll
  for (int k = d; k < C_DIM; k += 32) {
    const float qv = qrow[k];
    sx += qv * wrx[k];
    sy += qv * wry[k];
  }
  for (int off = 16; off > 0; off >>= 1) {
    sx += __shfl_xor(sx, off, 32);
    sy += __shfl_xor(sy, off, 32);
  }
  const float px = refpts[(size_t)i * 2 + 0] + sx + boff[2 * h];
  const float py = refpts[(size_t)i * 2 + 1] + sy + boff[2 * h + 1];

  // bilinear coords (grid_sample, zeros padding, align_corners=False)
  const float fx = px * (float)Wf - 0.5f;
  const float fy = py * (float)Hf - 0.5f;
  const float x0f = floorf(fx), y0f = floorf(fy);
  const float wx = fx - x0f, wy = fy - y0f;
  const int x0 = (int)x0f, y0 = (int)y0f;
  const int x1 = x0 + 1,   y1 = y0 + 1;

  float ks = 0.f, vs = 0.f;
  const size_t base = ((size_t)n * L) * C_DIM + (size_t)h * HDIM + d;
  auto gather = [&](int yi, int xi, float wgt) {
    if (yi >= 0 && yi < Hf && xi >= 0 && xi < Wf) {
      const size_t idx = base + ((size_t)yi * Wf + xi) * C_DIM;
      ks += wgt * K[idx];
      vs += wgt * V[idx];
    }
  };
  gather(y0, x0, (1.f - wy) * (1.f - wx));
  gather(y0, x1, (1.f - wy) * wx);
  gather(y1, x0, wy * (1.f - wx));
  gather(y1, x1, wy * wx);

  // head dot: wave-reduce q . k_s, scale 1/sqrt(32)
  float ap = Q[(size_t)i * C_DIM + tid] * ks;
  for (int off = 16; off > 0; off >>= 1) ap += __shfl_xor(ap, off, 32);
  if (d == 0) alog[h] = ap * 0.17677669529663689f;
  __syncthreads();

  // softmax over the 8 heads
  float mx = alog[0];
#pragma unroll
  for (int j = 1; j < HEADS; ++j) mx = fmaxf(mx, alog[j]);
  float denom = 0.f;
#pragma unroll
  for (int j = 0; j < HEADS; ++j) denom += __expf(alog[j] - mx);
  const float wsoft = __expf(alog[h] - mx) / denom;

  attn_bf[(size_t)i * C_DIM + tid] = f32_to_bf16_bits(wsoft * vs);
}

// --------------------------------------------------------------------------
extern "C" void kernel_launch(void* const* d_in, const int* in_sizes, int n_in,
                              void* d_out, int out_size, void* d_ws, size_t ws_size,
                              hipStream_t stream) {
  const float* query  = (const float*)d_in[0];
  const float* key    = (const float*)d_in[1];
  const float* value  = (const float*)d_in[2];
  const float* refpts = (const float*)d_in[3];
  const float* pos    = (const float*)d_in[4];
  const float* Wq = (const float*)d_in[5];   const float* bq = (const float*)d_in[6];
  const float* Wk = (const float*)d_in[7];   const float* bk = (const float*)d_in[8];
  const float* Wv = (const float*)d_in[9];   const float* bv = (const float*)d_in[10];
  const float* Wp = (const float*)d_in[11];  const float* bp = (const float*)d_in[12];
  const float* Woff = (const float*)d_in[13]; const float* boff = (const float*)d_in[14];
  const float* Wout = (const float*)d_in[15]; const float* bout = (const float*)d_in[16];
  const int* pHf = (const int*)d_in[17];
  const int* pWf = (const int*)d_in[18];

  const int M = in_sizes[0] / C_DIM;   // N*L = 36864

  // ---- scratch layout ----
  char* ws = (char*)d_ws;
  auto take = [&](size_t bytes) -> void* {
    void* p = (void*)ws;
    ws += (bytes + 255) & ~(size_t)255;
    return p;
  };
  const size_t wbytes = (size_t)C_DIM * C_DIM * sizeof(unsigned short);
  unsigned short* Wq_bf   = (unsigned short*)take(wbytes);
  unsigned short* Wk_bf   = (unsigned short*)take(wbytes);
  unsigned short* Wv_bf   = (unsigned short*)take(wbytes);
  unsigned short* Wp_bf   = (unsigned short*)take(wbytes);
  unsigned short* Wout_bf = (unsigned short*)take(wbytes);
  const size_t hbytes = (size_t)M * C_DIM * sizeof(unsigned short);
  unsigned short* query_bf = (unsigned short*)take(hbytes);
  unsigned short* key_bf   = (unsigned short*)take(hbytes);
  unsigned short* value_bf = (unsigned short*)take(hbytes);
  unsigned short* pos_bf   = (unsigned short*)take(hbytes);
  unsigned short* attn_bf  = (unsigned short*)take(hbytes);
  const size_t fbytes = (size_t)M * C_DIM * sizeof(float);
  float* Qb = (float*)take(fbytes);
  float* Kb = (float*)take(fbytes);
  float* Vb = (float*)take(fbytes);

  // ---- 1) f32 -> bf16 conversions (weights + activations) ----
  const int w8 = C_DIM * C_DIM / 8;
  dim3 cw((w8 + 255) / 256), cb(256);
  convert_f32_bf16<<<cw, cb, 0, stream>>>(Wq,   Wq_bf,   w8);
  convert_f32_bf16<<<cw, cb, 0, stream>>>(Wk,   Wk_bf,   w8);
  convert_f32_bf16<<<cw, cb, 0, stream>>>(Wv,   Wv_bf,   w8);
  convert_f32_bf16<<<cw, cb, 0, stream>>>(Wp,   Wp_bf,   w8);
  convert_f32_bf16<<<cw, cb, 0, stream>>>(Wout, Wout_bf, w8);
  const int a8 = M * C_DIM / 8;
  dim3 ca((a8 + 255) / 256);
  convert_f32_bf16<<<ca, cb, 0, stream>>>(query, query_bf, a8);
  convert_f32_bf16<<<ca, cb, 0, stream>>>(key,   key_bf,   a8);
  convert_f32_bf16<<<ca, cb, 0, stream>>>(value, value_bf, a8);
  convert_f32_bf16<<<ca, cb, 0, stream>>>(pos,   pos_bf,   a8);

  // ---- 2) projection GEMMs (pos projection fused into Q and K) ----
  dim3 ggrid(M / 64, C_DIM / 128), gblk(256);
  gemm_bias_wmma<<<ggrid, gblk, 0, stream>>>(query_bf, Wq_bf, bq,
                                             pos_bf, Wp_bf, bp, Qb, M);
  gemm_bias_wmma<<<ggrid, gblk, 0, stream>>>(key_bf, Wk_bf, bk,
                                             pos_bf, Wp_bf, bp, Kb, M);
  gemm_bias_wmma<<<ggrid, gblk, 0, stream>>>(value_bf, Wv_bf, bv,
                                             nullptr, nullptr, nullptr, Vb, M);

  // ---- 3) offsets + bilinear gather + head-softmax attention ----
  attn_sample_kernel<<<M, 256, 0, stream>>>(Qb, Kb, Vb, query, Woff, boff,
                                            refpts, attn_bf, pHf, pWf);

  // ---- 4) output projection straight into d_out ----
  gemm_bias_wmma<<<ggrid, gblk, 0, stream>>>(attn_bf, Wout_bf, bout,
                                             nullptr, nullptr, nullptr,
                                             (float*)d_out, M);
}
